// MetaSR_27230092657031
// MI455X (gfx1250) — matrix operand order repaired and checked by reading, not compiled, not run on hardware
//
#include <hip/hip_runtime.h>
#include <hip/hip_bf16.h>

typedef __attribute__((ext_vector_type(16))) _Float16 v16h;
typedef __attribute__((ext_vector_type(8)))  float    v8f;

#define Bn    4
#define Cn    64
#define Hn    64
#define Wn    64
#define Qn    16384           // QH*QW = 128*128
#define HIDn  256
#define CK    576             // C*9
#define CKP   577             // padded LDS row stride (bank rotate)
#define OUTC  1728            // CK*3
#define NCH   108             // OUTC/16 column chunks
#define KCH   8               // HID/32 k chunks
#define W2P_ELEMS (OUTC*HIDn) // 442368 halfs, 884736 B in workspace

// ---------------------------------------------------------------------------
// Pre-pass: convert w2 (HID x OUTC, f32 row-major) to f16, pre-swizzled into
// the WMMA B-fragment layout:
//   w2p[((cc*8 + kc)*32 + lane)*16 + j] = (f16) w2[h][c]
//   with  c = cc*16 + (lane&15),  h = kc*32 + ((lane>>4)<<3) + j + (j>=8 ? 8 : 0)
// so each lane's v16h fragment is a contiguous, 32B-aligned 32-byte load.
// ---------------------------------------------------------------------------
__global__ void metasr_pack_w2(const float* __restrict__ w2,
                               _Float16* __restrict__ w2p) {
  int t = blockIdx.x * 256 + threadIdx.x;
  if (t >= W2P_ELEMS) return;
  int j    = t & 15;
  int lane = (t >> 4) & 31;
  int kc   = (t >> 9) & 7;
  int cc   = t >> 12;
  int h = kc * 32 + ((lane >> 4) << 3) + j + ((j >= 8) ? 8 : 0);
  int c = cc * 16 + (lane & 15);
  w2p[t] = (_Float16)w2[h * OUTC + c];
}

// ---------------------------------------------------------------------------
// Main fused kernel: one wave32 per 16-query tile.
//   1) geometry -> (rel_y, rel_x, r_rev) + nearest LR index per query row
//   2) stage unfolded q_feat (16 x 576) in LDS
//   3) MLP1 computed straight into WMMA A fragments (f16, 8 x v16h)
//   4) 108 chunks: 8 x v_wmma_f32_16x16x32_f16 (K=256), fused epilogue
//      pred[q,o] += q_feat[q,c/3] * (PW[q,c] + b2[c]) for this lane's column
//   5) ds_add_f32 reduction over lanes, write (16 x 3) f32 outputs
// ---------------------------------------------------------------------------
__launch_bounds__(32)
__global__ void metasr_main(const float* __restrict__ feat,
                            const float* __restrict__ coord,
                            const float* __restrict__ cell,
                            const float* __restrict__ w1,
                            const float* __restrict__ b1,
                            const _Float16* __restrict__ w2p,
                            const float* __restrict__ b2,
                            float* __restrict__ out) {
  __shared__ float qf[16 * CKP];      // 36928 B  staged unfolded features
  __shared__ float w1s[3 * HIDn];     //  3072 B
  __shared__ float b1s[HIDn];         //  1024 B
  __shared__ int   iys[16], ixs[16];
  __shared__ float reds[48];          // pred[row][o] reduction

  const int lane = threadIdx.x;       // 0..31
  const int tile = blockIdx.x;        // 0..4095  (B*Q/16)
  const int b    = tile >> 10;        // Q/16 = 1024 tiles per batch
  const int q0   = (tile & 1023) << 4;
  const int mrow = lane & 15;         // A-fragment row owned by this lane
  const int half = lane >> 4;

  // ---- stage MLP1 weights ----
  for (int i = lane; i < 3 * HIDn; i += 32) w1s[i] = w1[i];
  for (int i = lane; i < HIDn;     i += 32) b1s[i] = b1[i];
  for (int i = lane; i < 48;       i += 32) reds[i] = 0.0f;

  // ---- per-query geometry (lanes l and l+16 duplicate row l) ----
  const long cidx = ((long)b * Qn + (q0 + mrow)) * 2;
  const float cy = coord[cidx],  cx = coord[cidx + 1];
  const float ly = cell[cidx],   lx = cell[cidx + 1];
  const float coy = cy - 0.5f * ly;
  const float cox = cx - 0.5f * lx;
  const float EPSf = 1e-6f;
  float cqy = fminf(fmaxf(coy + EPSf, -1.0f + EPSf), 1.0f - EPSf);
  float cqx = fminf(fmaxf(cox + EPSf, -1.0f + EPSf), 1.0f - EPSf);
  // ix = clip(round(((x+1)*n - 1)/2), 0, n-1); rintf == round-half-even
  int iy = (int)fminf(fmaxf(rintf(((cqy + 1.0f) * (float)Hn - 1.0f) * 0.5f), 0.0f), 63.0f);
  int ix = (int)fminf(fmaxf(rintf(((cqx + 1.0f) * (float)Wn - 1.0f) * 0.5f), 0.0f), 63.0f);
  const float rely = (coy - ((float)iy * 0.03125f - 1.0f)) * 32.0f;
  const float relx = (cox - ((float)ix * 0.03125f - 1.0f)) * 32.0f;
  const float rrev = ly * 32.0f;
  if (half == 0) { iys[mrow] = iy; ixs[mrow] = ix; }
  __syncthreads();

  // ---- stage q_feat: 3x3 zero-padded unfold at nearest LR pixel ----
  const float* fb = feat + (long)b * (Cn * Hn * Wn);
  for (int r = 0; r < 16; ++r) {
    const int iyr = iys[r], ixr = ixs[r];
    for (int k = lane; k < CK; k += 32) {
      int chan = k / 9;
      int tap  = k - chan * 9;
      int ky   = tap / 3;
      int y = iyr + ky - 1;
      int x = ixr + (tap - ky * 3) - 1;
      float v = 0.0f;
      if ((unsigned)y < (unsigned)Hn && (unsigned)x < (unsigned)Wn)
        v = fb[(chan << 12) + (y << 6) + x];
      qf[r * CKP + k] = v;
    }
  }
  __syncthreads();

  // ---- MLP1 -> A fragments in the documented 16-bit A layout ----
  // lanes 0-15 hold K 0..7 / 16..23, lanes 16-31 hold K 8..15 / 24..31
  v16h afrag[KCH];
#pragma unroll
  for (int kc = 0; kc < KCH; ++kc) {
    v16h a;
#pragma unroll
    for (int j = 0; j < 16; ++j) {
      int h = kc * 32 + half * 8 + j + ((j >= 8) ? 8 : 0);
      float hv = rely * w1s[h] + relx * w1s[HIDn + h] + rrev * w1s[2 * HIDn + h] + b1s[h];
      a[j] = (_Float16)fmaxf(hv, 0.0f);
    }
    afrag[kc] = a;
  }

  // ---- main GEMM + fused per-column epilogue ----
  float part0[8] = {0,0,0,0,0,0,0,0};
  float part1[8] = {0,0,0,0,0,0,0,0};
  float part2[8] = {0,0,0,0,0,0,0,0};
  const float* qbase = qf + half * 8 * CKP;   // this lane's 8 C/D rows

  for (int cc = 0; cc < NCH; ++cc) {
    const _Float16* bp = w2p + ((size_t)(cc * KCH) * 32 + lane) * 16;
    __builtin_prefetch(bp + KCH * 512, 0, 1);  // next chunk -> global_prefetch
    v8f acc = {};
#pragma unroll
    for (int kc = 0; kc < KCH; ++kc) {
      v16h bf = *(const v16h*)(bp + kc * 512);
      acc = __builtin_amdgcn_wmma_f32_16x16x32_f16(
          /*neg_a=*/false, afrag[kc], /*neg_b=*/false, bf,
          /*c_mod=*/(short)0, acc, /*reuse_a=*/false, /*reuse_b=*/false);
    }
    // this lane owns output column c for all 8 of its C/D rows
    const int c = cc * 16 + mrow;
    const int k = c / 3;
    const int o = c - k * 3;
    const float b2v = b2[c];
    const float m0 = (o == 0) ? 1.0f : 0.0f;
    const float m1 = (o == 1) ? 1.0f : 0.0f;
    const float m2 = (o == 2) ? 1.0f : 0.0f;
    const float* qk = qbase + k;
#pragma unroll
    for (int i = 0; i < 8; ++i) {
      float t = qk[i * CKP] * (acc[i] + b2v);
      part0[i] += m0 * t;
      part1[i] += m1 * t;
      part2[i] += m2 * t;
    }
  }

  // ---- reduce 16 lanes per (row,o) via LDS float atomics ----
#pragma unroll
  for (int i = 0; i < 8; ++i) {
    int r = i + half * 8;
    atomicAdd(&reds[r * 3 + 0], part0[i]);
    atomicAdd(&reds[r * 3 + 1], part1[i]);
    atomicAdd(&reds[r * 3 + 2], part2[i]);
  }
  __syncthreads();

  float* ob = out + ((long)b * Qn + q0) * 3;
  for (int i = lane; i < 48; i += 32) ob[i] = reds[i];
}

// ---------------------------------------------------------------------------
extern "C" void kernel_launch(void* const* d_in, const int* in_sizes, int n_in,
                              void* d_out, int out_size, void* d_ws, size_t ws_size,
                              hipStream_t stream) {
  const float* feat  = (const float*)d_in[0];
  const float* coord = (const float*)d_in[1];
  const float* cell  = (const float*)d_in[2];
  const float* w1    = (const float*)d_in[3];
  const float* b1    = (const float*)d_in[4];
  const float* w2    = (const float*)d_in[5];
  const float* b2    = (const float*)d_in[6];
  float* out = (float*)d_out;

  _Float16* w2p = (_Float16*)d_ws;  // 884736 bytes of scratch

  metasr_pack_w2<<<(W2P_ELEMS + 255) / 256, 256, 0, stream>>>(w2, w2p);
  metasr_main<<<(Bn * Qn) / 16, 32, 0, stream>>>(feat, coord, cell, w1, b1,
                                                 w2p, b2, out);
}
